// EGNNModel_56066503082238
// MI455X (gfx1250) — matrix operand order repaired and checked by reading, not compiled, not run on hardware
//
#include <hip/hip_runtime.h>

// ---------------------------------------------------------------------------
// EGNN (3 stacks x 4 GCL) for MI455X / gfx1250: bf16 WMMA matmuls, f32 accum,
// atomic segment-sums, wave32 tiling (1 wave = 16 edges / 16 nodes).
// ---------------------------------------------------------------------------

typedef __bf16 bf16_t;
typedef __attribute__((ext_vector_type(16))) __bf16 v16bf;
typedef __attribute__((ext_vector_type(8)))  __bf16 v8bf;
typedef __attribute__((ext_vector_type(8)))  float  v8f;

#define NNODE  20000
#define NEDGE  640000
#define HID    128
#define NGRAPH 32

// ------------------------------- device helpers ----------------------------

__device__ __forceinline__ float silu_f(float x) { return x / (1.0f + __expf(-x)); }
__device__ __forceinline__ float sigm_f(float x) { return 1.0f / (1.0f + __expf(-x)); }

// Intra-wave LDS ordering fence: LDS ops from one wave are processed in order;
// waiting DScnt==0 guarantees prior ds_stores are visible to later ds_loads
// from any lane of this wave.
__device__ __forceinline__ void wave_fence() {
  __builtin_amdgcn_wave_barrier();
  asm volatile("s_wait_dscnt 0" ::: "memory");
  __builtin_amdgcn_wave_barrier();
}

__device__ __forceinline__ v8f wmma_bf16(v16bf a, v16bf b, v8f c) {
  return __builtin_amdgcn_wmma_f32_16x16x32_bf16(
      /*neg_a=*/false, a, /*neg_b=*/false, b,
      /*c_mod=*/(short)0, c, /*reuse_a=*/false, /*reuse_b=*/false);
}

// A fragment (16x32 bf16) from an LDS row-major [16][ldk] buffer.
// ISA layout: lanes 0-15 = rows M=0..15 holding K-halves {k0..k0+7, k0+16..k0+23},
// lanes 16-31 = same rows holding {k0+8..k0+15, k0+24..k0+31}.
__device__ __forceinline__ v16bf load_afrag(const bf16_t* buf, int ldk, int k0, int lane) {
  int m = lane & 15, g = (lane >> 4) & 1;
  const v8bf lo = *reinterpret_cast<const v8bf*>(buf + m * ldk + k0 + g * 8);
  const v8bf hi = *reinterpret_cast<const v8bf*>(buf + m * ldk + k0 + 16 + g * 8);
  v16bf a;
#pragma unroll
  for (int i = 0; i < 8; ++i) { a[i] = lo[i]; a[8 + i] = hi[i]; }
  return a;
}

// B fragment (32x16 bf16) from a pre-swizzled weight buffer.
// Layout per tile (kt,nt): lane L holds column n = L&15, K-half = L>>4,
// element i -> k = (L>>4)*8 + (i&7) + (i>>3)*16.  32 contiguous bytes per lane.
__device__ __forceinline__ v16bf load_bfrag(const bf16_t* W, int nT, int kt, int nt, int lane) {
  const bf16_t* p = W + ((((size_t)(kt * nT + nt)) * 32 + lane) << 4);
  return *reinterpret_cast<const v16bf*>(p);
}

// -------------------------- weight pre-swizzle -----------------------------

__global__ void k_swizzle(const float* __restrict__ W, int K, int N, int Kp,
                          bf16_t* __restrict__ dst) {
  int idx = blockIdx.x * 256 + threadIdx.x;
  if (idx >= Kp * N) return;
  int i    = idx & 15;
  int lane = (idx >> 4) & 31;
  int t    = idx >> 9;
  int nT   = N >> 4;
  int nt   = t % nT;
  int kt   = t / nT;
  int k = kt * 32 + ((lane >> 4) << 3) + (i & 7) + ((i >> 3) << 4);
  int n = nt * 16 + (lane & 15);
  dst[idx] = (k < K) ? (bf16_t)W[(size_t)k * N + n] : (bf16_t)0.0f;
}

// ------------------------------ small kernels ------------------------------

__global__ void k_deg(const int* __restrict__ erow, int E, float* __restrict__ cnt) {
  int e = blockIdx.x * blockDim.x + threadIdx.x;
  if (e < E) atomicAdd(&cnt[erow[e]], 1.0f);
}

__global__ void k_gdeg(const int* __restrict__ batch, int Nn, float* __restrict__ gcnt) {
  int n = blockIdx.x * blockDim.x + threadIdx.x;
  if (n < Nn) atomicAdd(&gcnt[batch[n]], 1.0f);
}

__global__ void k_coord(float* __restrict__ coord, float* __restrict__ trans,
                        const float* __restrict__ cnt, int Nn) {
  int idx = blockIdx.x * blockDim.x + threadIdx.x;
  if (idx >= Nn * 3) return;
  int n = idx / 3;
  coord[idx] += trans[idx] / fmaxf(cnt[n], 1.0f);
  trans[idx] = 0.0f;
}

__global__ void k_pool(const float* __restrict__ h, const int* __restrict__ batch,
                       int Nn, int F, float* __restrict__ pool) {
  int idx = blockIdx.x * blockDim.x + threadIdx.x;
  if (idx >= Nn * F) return;
  int n = idx / F, f = idx - n * F;
  atomicAdd(&pool[batch[n] * F + f], h[idx]);
}

// seg-mean -> batchnorm(train stats) -> relu -> write into "once" columns
__global__ void k_bnpool(const float* __restrict__ pool, const float* __restrict__ gcnt,
                         const float* __restrict__ gamma, const float* __restrict__ beta,
                         int F, float* __restrict__ once, int colOff, int onceLd) {
  int f = blockIdx.x * blockDim.x + threadIdx.x;
  if (f >= F) return;
  float vals[NGRAPH];
  float mu = 0.0f;
#pragma unroll
  for (int g = 0; g < NGRAPH; ++g) {
    float v = pool[g * F + f] / fmaxf(gcnt[g], 1.0f);
    vals[g] = v; mu += v;
  }
  mu *= (1.0f / NGRAPH);
  float var = 0.0f;
#pragma unroll
  for (int g = 0; g < NGRAPH; ++g) { float d = vals[g] - mu; var += d * d; }
  var *= (1.0f / NGRAPH);
  float is = rsqrtf(var + 1e-5f);
  float gm = gamma[f], bt = beta[f];
#pragma unroll
  for (int g = 0; g < NGRAPH; ++g)
    once[g * onceLd + colOff + f] = fmaxf(gm * (vals[g] - mu) * is + bt, 0.0f);
}

__global__ void k_fc1(const float* __restrict__ once, const float* __restrict__ W,
                      const float* __restrict__ b, float* __restrict__ out) {
  int idx = blockIdx.x * blockDim.x + threadIdx.x;
  if (idx >= NGRAPH * 178) return;
  int g = idx / 178, o = idx - g * 178;
  float s = b[o];
  for (int p = 0; p < 3; ++p)
    for (int c = 0; c < 224; ++c)
      s += once[g * 224 + c] * W[(p * 224 + c) * 178 + o];
  out[idx] = s;
}

__global__ void k_bnfc(float* __restrict__ io, const float* __restrict__ gamma,
                       const float* __restrict__ beta, int F) {
  int f = blockIdx.x * blockDim.x + threadIdx.x;
  if (f >= F) return;
  float vals[NGRAPH];
  float mu = 0.0f;
#pragma unroll
  for (int g = 0; g < NGRAPH; ++g) { vals[g] = io[g * F + f]; mu += vals[g]; }
  mu *= (1.0f / NGRAPH);
  float var = 0.0f;
#pragma unroll
  for (int g = 0; g < NGRAPH; ++g) { float d = vals[g] - mu; var += d * d; }
  var *= (1.0f / NGRAPH);
  float is = rsqrtf(var + 1e-5f);
  float gm = gamma[f], bt = beta[f];
#pragma unroll
  for (int g = 0; g < NGRAPH; ++g) io[g * F + f] = gm * (vals[g] - mu) * is + bt;
}

__global__ void k_fin(const float* __restrict__ in, const float* __restrict__ W,
                      const float* __restrict__ b, float* __restrict__ out) {
  int idx = blockIdx.x * blockDim.x + threadIdx.x;
  if (idx >= NGRAPH * 128) return;
  int g = idx / 128, o = idx - g * 128;
  float s = b[o];
  for (int c = 0; c < 178; ++c) s += in[g * 178 + c] * W[c * 128 + o];
  out[idx] = s;
}

// ---------------------------- WMMA GEMM (rows) -----------------------------
// C[16-row tile][Nout] = A(f32)[16][K] @ Wswz + bias ; optional bf16 mirror.
__global__ __launch_bounds__(128) void k_gemm(const float* __restrict__ A, int M, int K,
                                              const bf16_t* __restrict__ Ws,
                                              const float* __restrict__ bias, int Nout,
                                              float* __restrict__ Cf, bf16_t* __restrict__ Cb) {
  __shared__ bf16_t sG[4][16 * 128];
  int lane = threadIdx.x & 31, wv = threadIdx.x >> 5;
  int tile = blockIdx.x * 4 + wv;
  if (tile * 16 >= M) return;
  bf16_t* buf = sG[wv];
  int ln = lane & 15, g = lane >> 4, r0 = tile * 16;

  for (int idx = lane; idx < 16 * K; idx += 32) {
    int m = idx / K, k = idx - m * K;
    buf[m * K + k] = (bf16_t)A[(size_t)(r0 + m) * K + k];
  }
  wave_fence();

  int nT = Nout >> 4, kT = K >> 5;
  for (int nb = 0; nb < nT; ++nb) {
    float bi = bias[nb * 16 + ln];
    v8f acc;
#pragma unroll
    for (int r = 0; r < 8; ++r) acc[r] = bi;
    for (int kt = 0; kt < kT; ++kt)
      acc = wmma_bf16(load_afrag(buf, K, kt * 32, lane), load_bfrag(Ws, nT, kt, nb, lane), acc);
#pragma unroll
    for (int r = 0; r < 8; ++r) {
      int m = r + 8 * g;
      size_t o = (size_t)(r0 + m) * Nout + nb * 16 + ln;
      Cf[o] = acc[r];
      if (Cb) Cb[o] = (bf16_t)acc[r];
    }
  }
}

// ------------------------------- edge kernel -------------------------------

struct GclEdgeArgs {
  const bf16_t* h;        // [N][128] bf16
  const float*  coord;    // [N][3]
  const int*    erow;
  const int*    ecol;
  const bf16_t* ew1s;     // swizzled 256x128
  const float*  ew1last;  // row 256 of ew1 (radial), f32[128]
  const float*  eb1;
  const bf16_t* ew2s;     // 128x128
  const float*  eb2;
  const float*  aw;       // f32[128]
  const float*  ab;       // f32[1]
  const bf16_t* cw1s;     // 128x128
  const float*  cb1;
  const float*  cw2;      // f32[128]
  float*        agg;      // [N][128] f32 accum
  float*        trans;    // [N][3]  f32 accum
  int           E;
};

__global__ __launch_bounds__(128) void k_edge(GclEdgeArgs A) {
  __shared__ bf16_t sA[4][16 * 128];
  __shared__ bf16_t sB[4][16 * 128];
  __shared__ float  sCD[4][16 * 4];
  __shared__ int    sRow[4][16];
  __shared__ int    sCol[4][16];

  int lane = threadIdx.x & 31, wv = threadIdx.x >> 5;
  int tile = blockIdx.x * 4 + wv;
  if (tile * 16 >= A.E) return;
  bf16_t* bufA = sA[wv];
  bf16_t* bufB = sB[wv];
  float*  cd   = sCD[wv];
  int*    srw  = sRow[wv];
  int*    scl  = sCol[wv];
  int ln = lane & 15, g = lane >> 4;

  if (lane < 16) {
    int e = tile * 16 + lane;
    int r = A.erow[e], c = A.ecol[e];
    srw[lane] = r; scl[lane] = c;
    float dx = A.coord[r * 3 + 0] - A.coord[c * 3 + 0];
    float dy = A.coord[r * 3 + 1] - A.coord[c * 3 + 1];
    float dz = A.coord[r * 3 + 2] - A.coord[c * 3 + 2];
    cd[lane * 4 + 0] = dx; cd[lane * 4 + 1] = dy; cd[lane * 4 + 2] = dz;
    cd[lane * 4 + 3] = dx * dx + dy * dy + dz * dz;
  }
  wave_fence();

  // Gather h[row] -> bufA, h[col] -> bufB (bf16 rows, coalesced b64 loads).
  for (int idx = lane; idx < 512; idx += 32) {
    int m = idx >> 5, q = idx & 31;
    ((unsigned long long*)bufA)[m * 32 + q] =
        ((const unsigned long long*)(A.h + (size_t)srw[m] * HID))[q];
    ((unsigned long long*)bufB)[m * 32 + q] =
        ((const unsigned long long*)(A.h + (size_t)scl[m] * HID))[q];
  }
  wave_fence();

  // ---- stage 1: m1 = silu([hrow|hcol|radial] @ ew1 + eb1) -----------------
  v8f acc[8];
#pragma unroll
  for (int nb = 0; nb < 8; ++nb) {
    int n = nb * 16 + ln;
    float bi = A.eb1[n];
    float wl = A.ew1last[n];
#pragma unroll
    for (int r = 0; r < 8; ++r) acc[nb][r] = bi + cd[(r + 8 * g) * 4 + 3] * wl;
  }
#pragma unroll
  for (int kt = 0; kt < 4; ++kt) {
    v16bf a = load_afrag(bufA, 128, kt * 32, lane);
#pragma unroll
    for (int nb = 0; nb < 8; ++nb)
      acc[nb] = wmma_bf16(a, load_bfrag(A.ew1s, 8, kt, nb, lane), acc[nb]);
  }
#pragma unroll
  for (int kt = 0; kt < 4; ++kt) {
    v16bf a = load_afrag(bufB, 128, kt * 32, lane);
#pragma unroll
    for (int nb = 0; nb < 8; ++nb)
      acc[nb] = wmma_bf16(a, load_bfrag(A.ew1s, 8, 4 + kt, nb, lane), acc[nb]);
  }
  wave_fence();
#pragma unroll
  for (int nb = 0; nb < 8; ++nb)
#pragma unroll
    for (int r = 0; r < 8; ++r)
      bufA[(r + 8 * g) * 128 + nb * 16 + ln] = (bf16_t)silu_f(acc[nb][r]);
  wave_fence();

  // ---- stage 2: m2 = silu(m1 @ ew2 + eb2); attention gate -----------------
#pragma unroll
  for (int nb = 0; nb < 8; ++nb) {
    float bi = A.eb2[nb * 16 + ln];
#pragma unroll
    for (int r = 0; r < 8; ++r) acc[nb][r] = bi;
  }
#pragma unroll
  for (int kt = 0; kt < 4; ++kt) {
    v16bf a = load_afrag(bufA, 128, kt * 32, lane);
#pragma unroll
    for (int nb = 0; nb < 8; ++nb)
      acc[nb] = wmma_bf16(a, load_bfrag(A.ew2s, 8, kt, nb, lane), acc[nb]);
  }
  float patt[8];
#pragma unroll
  for (int r = 0; r < 8; ++r) patt[r] = 0.0f;
#pragma unroll
  for (int nb = 0; nb < 8; ++nb) {
    float awn = A.aw[nb * 16 + ln];
#pragma unroll
    for (int r = 0; r < 8; ++r) {
      float v = silu_f(acc[nb][r]);
      acc[nb][r] = v;
      patt[r] += v * awn;
    }
  }
#pragma unroll
  for (int off = 1; off < 16; off <<= 1)
#pragma unroll
    for (int r = 0; r < 8; ++r) patt[r] += __shfl_xor(patt[r], off, 32);
  float ab0 = A.ab[0];
  float attv[8];
#pragma unroll
  for (int r = 0; r < 8; ++r) attv[r] = sigm_f(patt[r] + ab0);

  wave_fence();
  // gate, stash m (bf16) for cw1, scatter-add into agg
#pragma unroll
  for (int nb = 0; nb < 8; ++nb)
#pragma unroll
    for (int r = 0; r < 8; ++r) {
      float v = acc[nb][r] * attv[r];
      int m = r + 8 * g, n = nb * 16 + ln;
      bufB[m * 128 + n] = (bf16_t)v;
      atomicAdd(&A.agg[(size_t)srw[m] * HID + n], v);
    }
  wave_fence();

  // ---- stage 3: c = silu(m @ cw1 + cb1); t = tanh(c . cw2) ----------------
#pragma unroll
  for (int nb = 0; nb < 8; ++nb) {
    float bi = A.cb1[nb * 16 + ln];
#pragma unroll
    for (int r = 0; r < 8; ++r) acc[nb][r] = bi;
  }
#pragma unroll
  for (int kt = 0; kt < 4; ++kt) {
    v16bf a = load_afrag(bufB, 128, kt * 32, lane);
#pragma unroll
    for (int nb = 0; nb < 8; ++nb)
      acc[nb] = wmma_bf16(a, load_bfrag(A.cw1s, 8, kt, nb, lane), acc[nb]);
  }
  float pd[8];
#pragma unroll
  for (int r = 0; r < 8; ++r) pd[r] = 0.0f;
#pragma unroll
  for (int nb = 0; nb < 8; ++nb) {
    float cwn = A.cw2[nb * 16 + ln];
#pragma unroll
    for (int r = 0; r < 8; ++r) pd[r] += silu_f(acc[nb][r]) * cwn;
  }
#pragma unroll
  for (int off = 1; off < 16; off <<= 1)
#pragma unroll
    for (int r = 0; r < 8; ++r) pd[r] += __shfl_xor(pd[r], off, 32);
#pragma unroll
  for (int r = 0; r < 8; ++r) pd[r] = tanhf(pd[r]);

  if (ln < 3) {
#pragma unroll
    for (int r = 0; r < 8; ++r) {
      int m = r + 8 * g;
      atomicAdd(&A.trans[(size_t)srw[m] * 3 + ln], cd[m * 4 + ln] * pd[r]);
    }
  }
}

// ------------------------------- node kernel -------------------------------

struct GclNodeArgs {
  float*        hF;     // [N][128] f32
  bf16_t*       hB;     // [N][128] bf16 mirror
  float*        agg;    // [N][128] f32 (zeroed here after use)
  const bf16_t* nw1s;   // swizzled 256x128
  const float*  nb1;
  const bf16_t* nw2s;   // 128x128
  const float*  nb2;
  int           Nn;
};

__global__ __launch_bounds__(128) void k_node(GclNodeArgs A) {
  __shared__ bf16_t sN[4][16 * 256];
  int lane = threadIdx.x & 31, wv = threadIdx.x >> 5;
  int tile = blockIdx.x * 4 + wv;
  if (tile * 16 >= A.Nn) return;
  bf16_t* buf = sN[wv];
  int ln = lane & 15, g = lane >> 4;
  int n0 = tile * 16;

  // stage [h | agg] as bf16 rows of K=256
  for (int idx = lane; idx < 512; idx += 32) {
    int m = idx >> 5, q = idx & 31;
    ((unsigned long long*)buf)[m * 64 + q] =
        ((const unsigned long long*)(A.hB + (size_t)(n0 + m) * HID))[q];
  }
  for (int idx = lane; idx < 16 * 128; idx += 32) {
    int m = idx >> 7, k = idx & 127;
    buf[m * 256 + 128 + k] = (bf16_t)A.agg[(size_t)(n0 + m) * HID + k];
  }
  wave_fence();

  v8f acc[8];
#pragma unroll
  for (int nb = 0; nb < 8; ++nb) {
    float bi = A.nb1[nb * 16 + ln];
#pragma unroll
    for (int r = 0; r < 8; ++r) acc[nb][r] = bi;
  }
#pragma unroll
  for (int kt = 0; kt < 8; ++kt) {
    v16bf a = load_afrag(buf, 256, kt * 32, lane);
#pragma unroll
    for (int nb = 0; nb < 8; ++nb)
      acc[nb] = wmma_bf16(a, load_bfrag(A.nw1s, 8, kt, nb, lane), acc[nb]);
  }
  wave_fence();
#pragma unroll
  for (int nb = 0; nb < 8; ++nb)
#pragma unroll
    for (int r = 0; r < 8; ++r)
      buf[(r + 8 * g) * 128 + nb * 16 + ln] = (bf16_t)silu_f(acc[nb][r]);
  wave_fence();

#pragma unroll
  for (int nb = 0; nb < 8; ++nb) {
    float bi = A.nb2[nb * 16 + ln];
#pragma unroll
    for (int r = 0; r < 8; ++r) acc[nb][r] = bi;
  }
#pragma unroll
  for (int kt = 0; kt < 4; ++kt) {
    v16bf a = load_afrag(buf, 128, kt * 32, lane);
#pragma unroll
    for (int nb = 0; nb < 8; ++nb)
      acc[nb] = wmma_bf16(a, load_bfrag(A.nw2s, 8, kt, nb, lane), acc[nb]);
  }

  // residual + writeback + re-zero agg for next layer
#pragma unroll
  for (int nb = 0; nb < 8; ++nb)
#pragma unroll
    for (int r = 0; r < 8; ++r) {
      int m = r + 8 * g, n = nb * 16 + ln;
      size_t o = (size_t)(n0 + m) * HID + n;
      float v = A.hF[o] + acc[nb][r];
      A.hF[o] = v;
      A.hB[o] = (bf16_t)v;
      A.agg[o] = 0.0f;
    }
}

// -------------------------------- host side --------------------------------
// d_in: 0=x [20000,64], 1=coords [20000,3], 2=edge_index [2,640000] (int),
//       3=batch_ids [20000] (int), then 180 param leaves (pytree alpha order):
// 4:bn1_b 5:bn1_g 6:bn2_b 7:bn2_g 8:bn3_b 9:bn3_g
// egnn1 @10, egnn2 @66, egnn3 @122 : {emb_in_b, emb_in_w, emb_out_b, emb_out_w,
//   layers[l] @ base+4+13*l : {ab,aw,cb1,cw1,cw2,eb1,eb2,ew1,ew2,nb1,nb2,nw1,nw2}}
// 178:fc1_b 179:fc1_bn_b 180:fc1_bn_g 181:fc1_w 182:fin_b 183:fin_w

extern "C" void kernel_launch(void* const* d_in, const int* in_sizes, int n_in,
                              void* d_out, int out_size, void* d_ws, size_t ws_size,
                              hipStream_t stream) {
  (void)in_sizes; (void)n_in; (void)out_size; (void)ws_size;

  const float* x      = (const float*)d_in[0];
  const float* coords = (const float*)d_in[1];
  const int*   eidx   = (const int*)d_in[2];
  const int*   batch  = (const int*)d_in[3];
  const int*   erow   = eidx;
  const int*   ecol   = eidx + NEDGE;
  auto PF = [&](int i) -> const float* { return (const float*)d_in[i]; };

  // ---- workspace bump allocator (deterministic layout every call) ----
  char*  ws  = (char*)d_ws;
  size_t off = 0;
  auto alloc = [&](size_t bytes) -> char* {
    char* p = ws + off;
    off = (off + bytes + 255) & ~(size_t)255;
    return p;
  };
  float*  coordW = (float*)alloc((size_t)NNODE * 3 * 4);
  float*  cnt    = (float*)alloc((size_t)NNODE * 4);
  float*  gcnt   = (float*)alloc((size_t)NGRAPH * 4);
  float*  hF     = (float*)alloc((size_t)NNODE * HID * 4);
  bf16_t* hB     = (bf16_t*)alloc((size_t)NNODE * HID * 2);
  float*  agg    = (float*)alloc((size_t)NNODE * HID * 4);
  float*  trans  = (float*)alloc((size_t)NNODE * 3 * 4);
  float*  out1   = (float*)alloc((size_t)NNODE * 128 * 4);
  float*  out2   = (float*)alloc((size_t)NNODE * 64 * 4);
  float*  out3   = (float*)alloc((size_t)NNODE * 32 * 4);
  float*  pool1  = (float*)alloc((size_t)NGRAPH * 128 * 4);
  float*  pool2  = (float*)alloc((size_t)NGRAPH * 64 * 4);
  float*  pool3  = (float*)alloc((size_t)NGRAPH * 32 * 4);
  float*  once   = (float*)alloc((size_t)NGRAPH * 224 * 4);
  float*  fc1o   = (float*)alloc((size_t)NGRAPH * 178 * 4);

  auto swz = [&](const float* W, int K, int N) -> const bf16_t* {
    int Kp = (K + 31) & ~31;
    bf16_t* dst = (bf16_t*)alloc((size_t)Kp * N * 2);
    int total = Kp * N;
    k_swizzle<<<(total + 255) / 256, 256, 0, stream>>>(W, K, N, Kp, dst);
    return dst;
  };

  // ---- init ----
  hipMemsetAsync(cnt, 0, (size_t)NNODE * 4, stream);
  hipMemsetAsync(gcnt, 0, (size_t)NGRAPH * 4, stream);
  hipMemsetAsync(agg, 0, (size_t)NNODE * HID * 4, stream);
  hipMemsetAsync(trans, 0, (size_t)NNODE * 3 * 4, stream);
  hipMemsetAsync(pool1, 0, (size_t)NGRAPH * 128 * 4, stream);
  hipMemsetAsync(pool2, 0, (size_t)NGRAPH * 64 * 4, stream);
  hipMemsetAsync(pool3, 0, (size_t)NGRAPH * 32 * 4, stream);
  hipMemcpyAsync(coordW, coords, (size_t)NNODE * 3 * 4, hipMemcpyDeviceToDevice, stream);
  k_deg<<<(NEDGE + 255) / 256, 256, 0, stream>>>(erow, NEDGE, cnt);
  k_gdeg<<<(NNODE + 255) / 256, 256, 0, stream>>>(batch, NNODE, gcnt);

  const int    stackBase[3] = {10, 66, 122};
  const int    inK[3]       = {64, 128, 64};
  const int    outN[3]      = {128, 64, 32};
  const float* stackIn[3]   = {x, out1, out2};
  float*       stackOut[3]  = {out1, out2, out3};

  const int nodeBlocks = (NNODE / 16 + 3) / 4;   // 313
  const int edgeBlocks = (NEDGE / 16 + 3) / 4;   // 10000

  for (int s = 0; s < 3; ++s) {
    int b = stackBase[s];
    const bf16_t* einS = swz(PF(b + 1), inK[s], 128);
    k_gemm<<<nodeBlocks, 128, 0, stream>>>(stackIn[s], NNODE, inK[s], einS, PF(b + 0),
                                           128, hF, hB);
    for (int l = 0; l < 4; ++l) {
      int lb = b + 4 + 13 * l;
      GclEdgeArgs ea;
      ea.h = hB; ea.coord = coordW; ea.erow = erow; ea.ecol = ecol;
      ea.ew1s = swz(PF(lb + 7), 256, 128);
      ea.ew1last = PF(lb + 7) + 256 * 128;
      ea.eb1 = PF(lb + 5);
      ea.ew2s = swz(PF(lb + 8), 128, 128);
      ea.eb2 = PF(lb + 6);
      ea.aw = PF(lb + 1); ea.ab = PF(lb + 0);
      ea.cw1s = swz(PF(lb + 3), 128, 128);
      ea.cb1 = PF(lb + 2); ea.cw2 = PF(lb + 4);
      ea.agg = agg; ea.trans = trans; ea.E = NEDGE;
      k_edge<<<edgeBlocks, 128, 0, stream>>>(ea);
      k_coord<<<(NNODE * 3 + 255) / 256, 256, 0, stream>>>(coordW, trans, cnt, NNODE);
      GclNodeArgs na;
      na.hF = hF; na.hB = hB; na.agg = agg;
      na.nw1s = swz(PF(lb + 11), 256, 128); na.nb1 = PF(lb + 9);
      na.nw2s = swz(PF(lb + 12), 128, 128); na.nb2 = PF(lb + 10);
      na.Nn = NNODE;
      k_node<<<nodeBlocks, 128, 0, stream>>>(na);
    }
    const bf16_t* eoutS = swz(PF(b + 3), 128, outN[s]);
    k_gemm<<<nodeBlocks, 128, 0, stream>>>(hF, NNODE, 128, eoutS, PF(b + 2),
                                           outN[s], stackOut[s], nullptr);
  }

  // ---- pooling / BN / head ----
  k_pool<<<((size_t)NNODE * 128 + 255) / 256, 256, 0, stream>>>(out1, batch, NNODE, 128, pool1);
  k_pool<<<((size_t)NNODE * 64 + 255) / 256, 256, 0, stream>>>(out2, batch, NNODE, 64, pool2);
  k_pool<<<((size_t)NNODE * 32 + 255) / 256, 256, 0, stream>>>(out3, batch, NNODE, 32, pool3);
  k_bnpool<<<1, 128, 0, stream>>>(pool1, gcnt, PF(5), PF(4), 128, once, 0, 224);
  k_bnpool<<<1, 64, 0, stream>>>(pool2, gcnt, PF(7), PF(6), 64, once, 128, 224);
  k_bnpool<<<1, 32, 0, stream>>>(pool3, gcnt, PF(9), PF(8), 32, once, 192, 224);
  k_fc1<<<(NGRAPH * 178 + 255) / 256, 256, 0, stream>>>(once, PF(181), PF(178), fc1o);
  k_bnfc<<<1, 192, 0, stream>>>(fc1o, PF(180), PF(179), 178);
  k_fin<<<(NGRAPH * 128 + 255) / 256, 256, 0, stream>>>(fc1o, PF(183), PF(182), (float*)d_out);
}
